// RealGPTModel_53197464928558
// MI455X (gfx1250) — matrix operand order repaired and checked by reading, not compiled, not run on hardware
//
#include <hip/hip_runtime.h>
#include <math.h>

#define GPT_B  4
#define GPT_T  1024
#define GPT_C  768
#define GPT_L  12
#define GPT_H  12
#define GPT_FF 3072
#define GPT_V  50257
#define GPT_M  (GPT_B * GPT_T)   // 4096 rows in the flattened [B*T, C] activation

typedef __bf16 bf16;
typedef __attribute__((ext_vector_type(16))) __bf16 bf16x16;
typedef __attribute__((ext_vector_type(8)))  float  f32x8;

// Reinterpret two 16B chunks as one 16-element bf16 fragment
union FragU {
  uint4   v[2];
  bf16x16 f;
};

// D(16x16 f32) = A(16x32 bf16) * B(32x16 bf16) + C ; wave32, f32 accumulate
__device__ __forceinline__ f32x8 wmma_bf16f32(bf16x16 a, bf16x16 b, f32x8 c) {
  return __builtin_amdgcn_wmma_f32_16x16x32_bf16(false, a, false, b, (short)0, c,
                                                 false, false);
}

// CDNA5 async global->LDS copy (GVS addressing: SGPR base + 32-bit VGPR offset).
// lds_off = absolute LDS byte address (groupstaticsize + offset into dynamic LDS)
__device__ __forceinline__ void async_b128(unsigned lds_off, const void* base,
                                           unsigned byte_off) {
  asm volatile("global_load_async_to_lds_b128 %0, %1, %2"
               :: "v"(lds_off), "v"(byte_off), "s"(base) : "memory");
}
__device__ __forceinline__ void wait_async0() {
  asm volatile("s_wait_asynccnt 0x0" ::: "memory");
}

// ---------------------------------------------------------------------------
// Elementwise / LN / transpose kernels
// ---------------------------------------------------------------------------
__global__ __launch_bounds__(256) void k_embed(const int* __restrict__ ids,
                                               const float* __restrict__ tok,
                                               const float* __restrict__ pos,
                                               float* __restrict__ x) {
  size_t i = (size_t)blockIdx.x * 256 + threadIdx.x;
  if (i >= (size_t)GPT_M * GPT_C) return;
  int c  = (int)(i % GPT_C);
  int bt = (int)(i / GPT_C);
  int t  = bt % GPT_T;
  x[i] = tok[(size_t)ids[bt] * GPT_C + c] + pos[(size_t)t * GPT_C + c];
}

__global__ __launch_bounds__(256) void k_f32_to_bf16(const float* __restrict__ in,
                                                     bf16* __restrict__ out,
                                                     size_t n) {
  size_t i = (size_t)blockIdx.x * 256 + threadIdx.x;
  if (i < n) out[i] = (bf16)in[i];
}

// in: [K][N] f32 row-major  ->  out: [N][K] bf16 row-major (coalesced both ways)
__global__ __launch_bounds__(256) void k_transpose_bf16(const float* __restrict__ in,
                                                        bf16* __restrict__ out,
                                                        int K, int N) {
  __shared__ float tile[32][33];
  int k0 = blockIdx.y * 32, n0 = blockIdx.x * 32;
  int tx = threadIdx.x & 31, ty = threadIdx.x >> 5;  // 8 rows per pass
  for (int r = ty; r < 32; r += 8)
    tile[r][tx] = in[(size_t)(k0 + r) * N + n0 + tx];
  __syncthreads();
  for (int r = ty; r < 32; r += 8)
    out[(size_t)(n0 + r) * K + k0 + tx] = (bf16)tile[tx][r];
}

// One 256-thread block per row; C = 768. Output bf16 (GEMM A operand).
__global__ __launch_bounds__(256) void k_layernorm(const float* __restrict__ x,
                                                   const float* __restrict__ w,
                                                   const float* __restrict__ b,
                                                   bf16* __restrict__ out) {
  const int C = GPT_C;
  int row = blockIdx.x;
  int tid = threadIdx.x;
  __shared__ float rs[256];
  __shared__ float rs2[256];
  const float* xr = x + (size_t)row * C;
  float s = 0.f, s2 = 0.f;
  for (int c = tid; c < C; c += 256) {
    float v = xr[c];
    s += v; s2 += v * v;
  }
  rs[tid] = s; rs2[tid] = s2;
  __syncthreads();
  for (int off = 128; off > 0; off >>= 1) {
    if (tid < off) { rs[tid] += rs[tid + off]; rs2[tid] += rs2[tid + off]; }
    __syncthreads();
  }
  float mean = rs[0] / C;
  float var  = rs2[0] / C - mean * mean;
  float rstd = rsqrtf(var + 1e-5f);
  for (int c = tid; c < C; c += 256) {
    out[(size_t)row * C + c] = (bf16)((xr[c] - mean) * rstd * w[c] + b[c]);
  }
}

// ---------------------------------------------------------------------------
// Tiled WMMA GEMM:  C[M,N] = A[M,K](bf16,row-major) * Bt[N,K](bf16,row-major)^T
// Block tile 128x256, 8 waves (2 x 4), wave tile 64x64 = 4x4 WMMA frags
// (16 b128 LDS reads per 16 WMMAs). Global->LDS staging is double-buffered
// GLOBAL_LOAD_ASYNC_TO_LDS_B128 (ASYNCcnt), one barrier per K-step.
// EPI: 0 = f32 store (+bias), 1 = residual += (+bias),
//      2 = exact GELU -> bf16 (+bias), 3 = f32 store, no bias, N-guarded
// ---------------------------------------------------------------------------
#define G_APITCH 40               // bf16 elements per A/B LDS row (80B)
#define G_AELE   (128 * G_APITCH) // elements per A buffer
#define G_BELE   (256 * G_APITCH) // elements per B buffer
#define G_SMEM_BYTES ((2 * G_AELE + 2 * G_BELE) * 2)  // 61440 B

template <int EPI>
__global__ __launch_bounds__(256) void k_gemm(const bf16* __restrict__ A,
                                              const bf16* __restrict__ Bt,
                                              const float* __restrict__ bias,
                                              void* __restrict__ Cout,
                                              int N, int K) {
  extern __shared__ char smem[];
  bf16* sA = (bf16*)smem;                 // [2][128][40]
  bf16* sB = (bf16*)smem + 2 * G_AELE;    // [2][256][40]
  const unsigned ldsbase = __builtin_amdgcn_groupstaticsize();

  const int tid  = threadIdx.x;
  const int wv   = tid >> 5;
  const int lane = tid & 31;
  const int lr   = lane & 15;    // fragment lane index
  const int hi   = lane >> 4;    // half-wave index
  const int wm   = wv >> 2;      // 0..1 -> 64-row slice
  const int wn   = wv & 3;       // 0..3 -> 64-col slice
  const int m0   = blockIdx.y * 128;
  const int n0   = blockIdx.x * 256;

  // async staging coordinates
  const int arow = tid >> 1;            // 0..127, 2 threads/row
  const int acol = (tid & 1) * 16;      // 16 k each (2 x b128)
  const int brow = tid;                 // 0..255, 1 thread/row, 32 k (4 x b128)
  const int gnld = n0 + brow;

  auto issue_tile = [&](int k0, int buf) {
    unsigned la = ldsbase + (unsigned)(buf * G_AELE + arow * G_APITCH + acol) * 2;
    unsigned ga = (unsigned)(((size_t)(m0 + arow) * K + k0 + acol) * 2);
    async_b128(la,      A, ga);
    async_b128(la + 16, A, ga + 16);
    if (EPI != 3 || gnld < N) {
      unsigned lb = ldsbase +
                    (unsigned)(2 * G_AELE + buf * G_BELE + brow * G_APITCH) * 2;
      unsigned gb = (unsigned)(((size_t)gnld * K + k0) * 2);
      async_b128(lb,      Bt, gb);
      async_b128(lb + 16, Bt, gb + 16);
      async_b128(lb + 32, Bt, gb + 32);
      async_b128(lb + 48, Bt, gb + 48);
    }
  };

  f32x8 acc[4][4];
  const f32x8 zero = {0.f, 0.f, 0.f, 0.f, 0.f, 0.f, 0.f, 0.f};
#pragma unroll
  for (int mf = 0; mf < 4; ++mf)
#pragma unroll
    for (int nf = 0; nf < 4; ++nf) acc[mf][nf] = zero;

  issue_tile(0, 0);
  int cur = 0;
  for (int k0 = 0; k0 < K; k0 += 32) {
    wait_async0();      // this wave's async copies into buf `cur` done
    __syncthreads();    // everyone's copies done; prior compute on other buf done
    if (k0 + 32 < K) issue_tile(k0 + 32, cur ^ 1);

    const bf16* Asb = sA + cur * G_AELE;
    const bf16* Bsb = sB + cur * G_BELE;

    // A fragments: elements 0..7 = k hi*8.., 8..15 = k 16+hi*8.. (2 x b128)
    bf16x16 af[4];
#pragma unroll
    for (int mf = 0; mf < 4; ++mf) {
      int ar = wm * 64 + mf * 16 + lr;
      FragU u;
      u.v[0] = *(const uint4*)&Asb[ar * G_APITCH + hi * 8];
      u.v[1] = *(const uint4*)&Asb[ar * G_APITCH + hi * 8 + 16];
      af[mf] = u.f;
    }
#pragma unroll
    for (int nf = 0; nf < 4; ++nf) {
      int bc = wn * 64 + nf * 16 + lr;
      FragU u;
      u.v[0] = *(const uint4*)&Bsb[bc * G_APITCH + hi * 8];
      u.v[1] = *(const uint4*)&Bsb[bc * G_APITCH + hi * 8 + 16];
#pragma unroll
      for (int mf = 0; mf < 4; ++mf)
        acc[mf][nf] = wmma_bf16f32(af[mf], u.f, acc[mf][nf]);
    }
    cur ^= 1;
  }

  // Epilogue: C/D layout -> row = r + 8*hi, col = lr within each 16x16 frag
#pragma unroll
  for (int mf = 0; mf < 4; ++mf)
#pragma unroll
    for (int nf = 0; nf < 4; ++nf)
#pragma unroll
      for (int r = 0; r < 8; ++r) {
        int gm = m0 + wm * 64 + mf * 16 + r + hi * 8;
        int gn = n0 + wn * 64 + nf * 16 + lr;
        float v = acc[mf][nf][r];
        if (EPI != 3) v += bias[gn];
        size_t o = (size_t)gm * N + gn;
        if (EPI == 0) {
          ((float*)Cout)[o] = v;
        } else if (EPI == 1) {
          float* p = (float*)Cout + o;
          *p = *p + v;
        } else if (EPI == 2) {
          float g = 0.5f * v * (1.0f + erff(v * 0.70710678118f));
          ((bf16*)Cout)[o] = (bf16)g;
        } else {
          if (gn < N) ((float*)Cout)[o] = v;
        }
      }
}

// ---------------------------------------------------------------------------
// Flash attention (causal, bf16 WMMA, online softmax).
// Grid: (T/128, H, B); 8 waves/block; each wave owns 16 q rows.
// qkv layout: [B*T, 3C] f32 with q|k|v head slices; out: bf16 [B*T, C].
// K tile kept [kpos][d] (QK^T frag is d-contiguous); V tile kept TRANSPOSED
// [d][kpos] (P.V frag is kpos-contiguous) -> all fragment reads are b128.
// ---------------------------------------------------------------------------
__global__ __launch_bounds__(256) void k_attention(const float* __restrict__ qkv,
                                                   bf16* __restrict__ out) {
  const int C3 = 3 * GPT_C;
  const int b  = blockIdx.z;
  const int h  = blockIdx.y;
  const int qt = blockIdx.x;

  const int tid  = threadIdx.x;
  const int wv   = tid >> 5;
  const int lane = tid & 31;
  const int lr   = lane & 15;
  const int hi   = lane >> 4;
  const int qbase = qt * 128 + wv * 16;

  __shared__ bf16 Kt[32][72];        // [kpos][d], 144B row stride
  __shared__ bf16 VtT[64][40];       // [d][kpos], 80B row stride
  __shared__ bf16 Ps[8][16][40];     // per-wave P staging, 80B row stride

  // Q fragments for the two d-chunks (d 0..31 and 32..63), loaded once.
  bf16x16 aq0, aq1;
  {
    int row = qbase + lr;
    const float* qp = qkv + (size_t)(b * GPT_T + row) * C3 + h * 64;
#pragma unroll
    for (int j = 0; j < 8; ++j) {
      int k = ((j < 4) ? 0 : 16) + hi * 8 + 2 * (j & 3);
      aq0[2 * j]     = (bf16)qp[k];
      aq0[2 * j + 1] = (bf16)qp[k + 1];
      aq1[2 * j]     = (bf16)qp[32 + k];
      aq1[2 * j + 1] = (bf16)qp[32 + k + 1];
    }
  }

  f32x8 oacc[4];
  const f32x8 zero = {0.f, 0.f, 0.f, 0.f, 0.f, 0.f, 0.f, 0.f};
#pragma unroll
  for (int nf = 0; nf < 4; ++nf) oacc[nf] = zero;
  float mrow[8], lrow[8];
#pragma unroll
  for (int r = 0; r < 8; ++r) { mrow[r] = -INFINITY; lrow[r] = 0.f; }

  const float scale = 0.125f;     // 1/sqrt(64)
  const int nchunk = qt * 4 + 4;  // causal bound for this 128-row block

  for (int kc = 0; kc < nchunk; ++kc) {
    __syncthreads();
    // Stage K (natural) and V (transposed) 32x64 tiles, f32 -> bf16
    for (int i = tid; i < 32 * 64; i += 256) {
      int kp = i >> 6, d = i & 63;
      size_t base = (size_t)(b * GPT_T + kc * 32 + kp) * C3 + h * 64 + d;
      Kt[kp][d]  = (bf16)qkv[base + GPT_C];
      VtT[d][kp] = (bf16)qkv[base + 2 * GPT_C];
    }
    __syncthreads();

    if (kc * 32 > qbase + 15) continue;  // wave-uniform causal skip

    // S = Q * K^T for 32 key positions: 2 d-steps x 2 n-frags
    f32x8 s0 = zero, s1 = zero;
#pragma unroll
    for (int dstep = 0; dstep < 2; ++dstep) {
      bf16x16 aq = dstep ? aq1 : aq0;
#pragma unroll
      for (int nf = 0; nf < 2; ++nf) {
        int kpos  = nf * 16 + lr;           // B-frag column = key position
        int dbase = dstep * 32 + hi * 8;    // d hi*8.. and 16+hi*8..
        FragU u;
        u.v[0] = *(const uint4*)&Kt[kpos][dbase];
        u.v[1] = *(const uint4*)&Kt[kpos][dbase + 16];
        if (nf == 0) s0 = wmma_bf16f32(aq, u.f, s0);
        else         s1 = wmma_bf16f32(aq, u.f, s1);
      }
    }

    // Online softmax update, write P into per-wave LDS tile
#pragma unroll
    for (int r = 0; r < 8; ++r) {
      int grow = qbase + r + hi * 8;
      float v0 = s0[r] * scale, v1 = s1[r] * scale;
      int c0 = kc * 32 + lr, c1 = c0 + 16;
      if (c0 > grow) v0 = -INFINITY;
      if (c1 > grow) v1 = -INFINITY;
      float mx = fmaxf(v0, v1);
#pragma unroll
      for (int off = 1; off < 16; off <<= 1)
        mx = fmaxf(mx, __shfl_xor(mx, off, 32));
      float mnew = fmaxf(mrow[r], mx);
      float sf = expf(mrow[r] - mnew);
      float p0 = expf(v0 - mnew);
      float p1 = expf(v1 - mnew);
      float sum = p0 + p1;
#pragma unroll
      for (int off = 1; off < 16; off <<= 1)
        sum += __shfl_xor(sum, off, 32);
      lrow[r] = lrow[r] * sf + sum;
      mrow[r] = mnew;
#pragma unroll
      for (int nf = 0; nf < 4; ++nf) oacc[nf][r] *= sf;
      Ps[wv][r + hi * 8][lr]      = (bf16)p0;
      Ps[wv][r + hi * 8][lr + 16] = (bf16)p1;
    }

    // Re-fragment P as A matrix (same-wave LDS roundtrip; DS ops are in-order)
    FragU up;
    up.v[0] = *(const uint4*)&Ps[wv][lr][hi * 8];
    up.v[1] = *(const uint4*)&Ps[wv][lr][hi * 8 + 16];
    // O += P * V
#pragma unroll
    for (int nf = 0; nf < 4; ++nf) {
      int dcol = nf * 16 + lr;
      FragU uv;
      uv.v[0] = *(const uint4*)&VtT[dcol][hi * 8];
      uv.v[1] = *(const uint4*)&VtT[dcol][hi * 8 + 16];
      oacc[nf] = wmma_bf16f32(up.f, uv.f, oacc[nf]);
    }
  }

  // Finalize: O /= l, store bf16 into [B*T, C] at head offset
#pragma unroll
  for (int r = 0; r < 8; ++r) {
    float inv = 1.0f / lrow[r];
    int grow = qbase + r + hi * 8;
    size_t orow = (size_t)(b * GPT_T + grow) * GPT_C + h * 64;
#pragma unroll
    for (int nf = 0; nf < 4; ++nf)
      out[orow + nf * 16 + lr] = (bf16)(oacc[nf][r] * inv);
  }
}

// ---------------------------------------------------------------------------
// Host orchestration
// ---------------------------------------------------------------------------
extern "C" void kernel_launch(void* const* d_in, const int* in_sizes, int n_in,
                              void* d_out, int out_size, void* d_ws, size_t ws_size,
                              hipStream_t stream) {
  const int C = GPT_C, FF = GPT_FF, V = GPT_V, M = GPT_M;
  const int* ids      = (const int*)d_in[0];
  const float* tok    = (const float*)d_in[1];
  const float* pos    = (const float*)d_in[2];
  const float* qkv_w  = (const float*)d_in[3];
  const float* qkv_b  = (const float*)d_in[4];
  const float* proj_w = (const float*)d_in[5];
  const float* proj_b = (const float*)d_in[6];
  const float* ln1_w  = (const float*)d_in[7];
  const float* ln1_b  = (const float*)d_in[8];
  const float* ln2_w  = (const float*)d_in[9];
  const float* ln2_b  = (const float*)d_in[10];
  const float* fc1_w  = (const float*)d_in[11];
  const float* fc1_b  = (const float*)d_in[12];
  const float* fc2_w  = (const float*)d_in[13];
  const float* fc2_b  = (const float*)d_in[14];
  const float* lnf_w  = (const float*)d_in[15];
  const float* lnf_b  = (const float*)d_in[16];

  char* ws = (char*)d_ws;
  size_t off = 0;
  auto alloc = [&](size_t bytes) -> void* {
    void* p = ws + off;
    off = (off + bytes + 255) & ~(size_t)255;
    return p;
  };
  float* x      = (float*)alloc((size_t)M * C * 4);       // residual stream f32
  bf16*  hbuf   = (bf16*) alloc((size_t)M * C * 2);       // LN output bf16
  float* qkvbuf = (float*)alloc((size_t)M * 3 * C * 4);   // qkv f32
  bf16*  attno  = (bf16*) alloc((size_t)M * C * 2);       // attention out bf16
  bf16*  ffbuf  = (bf16*) alloc((size_t)M * FF * 2);      // GELU out bf16
  bf16*  wq     = (bf16*) alloc((size_t)C * 3 * C * 2);   // [3C][C]  bf16
  bf16*  wp     = (bf16*) alloc((size_t)C * C * 2);       // [C][C]   bf16
  bf16*  w1     = (bf16*) alloc((size_t)C * FF * 2);      // [FF][C]  bf16
  bf16*  w2     = (bf16*) alloc((size_t)FF * C * 2);      // [C][FF]  bf16
  bf16*  tokbf  = (bf16*) alloc((size_t)V * C * 2);       // [V][C]   bf16
  if (off > ws_size) return;  // workspace too small (~180 MB needed)

  auto blocks = [](size_t n) { return (unsigned)((n + 255) / 256); };

  // 1) embeddings
  k_embed<<<blocks((size_t)M * C), 256, 0, stream>>>(ids, tok, pos, x);

  // 2) transformer layers
  for (int l = 0; l < GPT_L; ++l) {
    size_t nq = (size_t)C * 3 * C, np = (size_t)C * C, nf = (size_t)C * FF;
    // weights stored [in(K), out(N)] f32 -> [N][K] bf16 (transposed)
    k_transpose_bf16<<<dim3(3 * C / 32, C / 32), 256, 0, stream>>>(
        qkv_w + (size_t)l * nq, wq, C, 3 * C);
    k_transpose_bf16<<<dim3(C / 32, C / 32), 256, 0, stream>>>(
        proj_w + (size_t)l * np, wp, C, C);
    k_transpose_bf16<<<dim3(FF / 32, C / 32), 256, 0, stream>>>(
        fc1_w + (size_t)l * nf, w1, C, FF);
    k_transpose_bf16<<<dim3(C / 32, FF / 32), 256, 0, stream>>>(
        fc2_w + (size_t)l * nf, w2, FF, C);

    k_layernorm<<<M, 256, 0, stream>>>(x, ln1_w + (size_t)l * C,
                                       ln1_b + (size_t)l * C, hbuf);
    k_gemm<0><<<dim3(3 * C / 256, M / 128), 256, G_SMEM_BYTES, stream>>>(
        hbuf, wq, qkv_b + (size_t)l * 3 * C, (void*)qkvbuf, 3 * C, C);

    k_attention<<<dim3(GPT_T / 128, GPT_H, GPT_B), 256, 0, stream>>>(qkvbuf, attno);

    k_gemm<1><<<dim3(C / 256, M / 128), 256, G_SMEM_BYTES, stream>>>(
        attno, wp, proj_b + (size_t)l * C, (void*)x, C, C);

    k_layernorm<<<M, 256, 0, stream>>>(x, ln2_w + (size_t)l * C,
                                       ln2_b + (size_t)l * C, hbuf);
    k_gemm<2><<<dim3(FF / 256, M / 128), 256, G_SMEM_BYTES, stream>>>(
        hbuf, w1, fc1_b + (size_t)l * FF, (void*)ffbuf, FF, C);
    k_gemm<1><<<dim3(C / 256, M / 128), 256, G_SMEM_BYTES, stream>>>(
        ffbuf, w2, fc2_b + (size_t)l * C, (void*)x, C, FF);
  }

  // 3) final LN + tied lm_head (tok_emb is already [N][K] = [V][C])
  k_layernorm<<<M, 256, 0, stream>>>(x, lnf_w, lnf_b, hbuf);
  k_f32_to_bf16<<<blocks((size_t)V * C), 256, 0, stream>>>(tok, tokbf,
                                                           (size_t)V * C);
  k_gemm<3><<<dim3((V + 255) / 256, M / 128), 256, G_SMEM_BYTES, stream>>>(
      hbuf, tokbf, nullptr, (void*)d_out, V, C);
}
